// GPT_75179107549988
// MI455X (gfx1250) — compile-verified, matrix operand order checked by reading, not run on hardware
//
#include <hip/hip_runtime.h>
#include <hip/hip_bf16.h>

typedef __attribute__((ext_vector_type(16))) _Float16 v16h;
typedef __attribute__((ext_vector_type(8)))  _Float16 v8h;
typedef __attribute__((ext_vector_type(8)))  float    v8f;

#define TQ    510
#define TP    512
#define BQ    8
#define EXD   768
#define NHD   12
#define HDD   64
#define VOC   256
#define NBLKC 11
#define HID   3072
#define ROWS  (BQ * TP)   // 4096
#define KTOT  (TQ * EXD)  // 391680 = 102 * 3840
#define NKB   102         // head k-split blocks
#define KPB   (KTOT / NKB) // 3840

// -------------------------------------------------------------------------
// Embedding: x[b,t] = relu(lin_w[tok] + lin_w[2+t] + lin_b); pad rows = 0
// -------------------------------------------------------------------------
__global__ __launch_bounds__(256) void emb_kernel(const int* __restrict__ idx,
                                                  const float* __restrict__ lin_w,
                                                  const float* __restrict__ lin_b,
                                                  float* __restrict__ X) {
  int row = blockIdx.x;            // 0..4095
  int b = row >> 9, t = row & 511;
  float* xr = X + (size_t)row * EXD;
  if (t >= TQ) {
    for (int j = threadIdx.x; j < EXD; j += 256) xr[j] = 0.0f;
    return;
  }
  int tok = idx[b * TQ + t];       // 0 or 1 -> one-hot row 0 or 1
  const float* wt = lin_w + (size_t)tok * EXD;
  const float* wp = lin_w + (size_t)(2 + t) * EXD;
  for (int j = threadIdx.x; j < EXD; j += 256) {
    float v = wt[j] + wp[j] + lin_b[j];
    xr[j] = v > 0.0f ? v : 0.0f;
  }
}

// -------------------------------------------------------------------------
// LayerNorm over 768 columns, one block (256 thr) per row, f32 out
// -------------------------------------------------------------------------
__global__ __launch_bounds__(256) void ln_kernel(const float* __restrict__ X,
                                                 const float* __restrict__ sc,
                                                 const float* __restrict__ bi,
                                                 float* __restrict__ O) {
  int row = blockIdx.x;
  const float* xr = X + (size_t)row * EXD;
  float* orow = O + (size_t)row * EXD;
  __shared__ float red[256];
  int tid = threadIdx.x;
  float v[3];
  float s = 0.0f;
#pragma unroll
  for (int i = 0; i < 3; ++i) { v[i] = xr[tid + i * 256]; s += v[i]; }
  red[tid] = s; __syncthreads();
  for (int o = 128; o > 0; o >>= 1) { if (tid < o) red[tid] += red[tid + o]; __syncthreads(); }
  float mean = red[0] * (1.0f / EXD);
  __syncthreads();
  float sq = 0.0f;
#pragma unroll
  for (int i = 0; i < 3; ++i) { float d = v[i] - mean; sq += d * d; }
  red[tid] = sq; __syncthreads();
  for (int o = 128; o > 0; o >>= 1) { if (tid < o) red[tid] += red[tid + o]; __syncthreads(); }
  float rstd = rsqrtf(red[0] * (1.0f / EXD) + 1e-5f);
#pragma unroll
  for (int i = 0; i < 3; ++i) {
    int j = tid + i * 256;
    orow[j] = (v[i] - mean) * rstd * sc[j] + bi[j];
  }
}

// -------------------------------------------------------------------------
// Fragment loader: the 16 f16 elements of an A/B WMMA operand for one lane
// are two contiguous 16-byte LDS chunks -> two ds_load_b128.
// -------------------------------------------------------------------------
__device__ __forceinline__ v16h frag16(const _Float16* p) {
  v8h lo = *(const v8h*)p;
  v8h hi = *(const v8h*)(p + 16);
  v16h r;
#pragma unroll
  for (int i = 0; i < 8; ++i) { r[i] = lo[i]; r[i + 8] = hi[i]; }
  return r;
}

// -------------------------------------------------------------------------
// Batched WMMA GEMM, tile 128x64x32, 8 waves, each wave owns 32x32 of C
// (4 x v_wmma_f32_16x16x32_f16 per K-step).
//   C[z][m,n] = op( alpha * sum_k A[z][m,k]*B[z][k,n] + bias[n] + Res[z][m,n] )
// A row-major [M,K]; BT=false: B row-major; BT=true: B K-contiguous.
// B staged TRANSPOSED in LDS ([n][k]) so B fragment gathers are b128 too.
// M % 128 == 0, N % 64 == 0, K % 32 == 0 -> no bounds checks, EXEC all-ones.
// -------------------------------------------------------------------------
template <bool BT>
__global__ __launch_bounds__(256) void gemm_wmma(
    const float* __restrict__ A, int lda, long bsAb, long bsAh,
    const float* __restrict__ Bm, int ldb, long bsBb, long bsBh,
    const float* __restrict__ bias,
    const float* __restrict__ Res, long bsRb, long bsRh,
    float* __restrict__ C, int ldc, long bsCb, long bsCh,
    int K, int nh, float alpha, int relu) {
  int z = blockIdx.z;
  int zb = z / nh, zh = z % nh;
  A  += (size_t)zb * bsAb + (size_t)zh * bsAh;
  Bm += (size_t)zb * bsBb + (size_t)zh * bsBh;
  C  += (size_t)zb * bsCb + (size_t)zh * bsCh;
  if (Res) Res += (size_t)zb * bsRb + (size_t)zh * bsRh;

  const int bm = blockIdx.y * 128;
  const int bn = blockIdx.x * 64;

  __shared__ alignas(16) _Float16 As[128][40];  // [m][k]
  __shared__ alignas(16) _Float16 Bt[64][40];   // [n][k] (transposed)

  const int tid = threadIdx.x;
  const int w  = tid >> 5, l = tid & 31;
  const int wm = w >> 1,  wn = w & 1;        // 4x2 wave grid (32x32 C/wave)
  const int lr = l & 15;
  const int hi = (l >> 4) & 1;

  const int ar = tid >> 1, ac = (tid & 1) * 16;
  const float* aPtr = A + (size_t)(bm + ar) * lda + ac;

  const int br  = tid >> 3, bc = (tid & 7) * 8;
  const int bnn = tid >> 2, ks = (tid & 3) * 8;
  const float* bPtr = BT ? (Bm + (size_t)(bn + bnn) * ldb + ks)
                         : (Bm + (size_t)br * ldb + bn + bc);
  const size_t bStep = BT ? 32 : (size_t)32 * ldb;

  v8f acc00 = {}, acc01 = {}, acc10 = {}, acc11 = {};

  for (int kt = 0; kt < K; kt += 32) {
    {
      const float4* p = (const float4*)aPtr;
      float4 u0 = p[0], u1 = p[1], u2 = p[2], u3 = p[3];
      v8h h0 = {(_Float16)u0.x, (_Float16)u0.y, (_Float16)u0.z, (_Float16)u0.w,
                (_Float16)u1.x, (_Float16)u1.y, (_Float16)u1.z, (_Float16)u1.w};
      v8h h1 = {(_Float16)u2.x, (_Float16)u2.y, (_Float16)u2.z, (_Float16)u2.w,
                (_Float16)u3.x, (_Float16)u3.y, (_Float16)u3.z, (_Float16)u3.w};
      *(v8h*)&As[ar][ac]     = h0;
      *(v8h*)&As[ar][ac + 8] = h1;
    }
    {
      const float4* p = (const float4*)bPtr;
      float4 u0 = p[0], u1 = p[1];
      _Float16 h[8] = {(_Float16)u0.x, (_Float16)u0.y, (_Float16)u0.z, (_Float16)u0.w,
                       (_Float16)u1.x, (_Float16)u1.y, (_Float16)u1.z, (_Float16)u1.w};
      if (BT) {
        v8h hv = {h[0], h[1], h[2], h[3], h[4], h[5], h[6], h[7]};
        *(v8h*)&Bt[bnn][ks] = hv;
      } else {
#pragma unroll
        for (int j = 0; j < 8; ++j) Bt[bc + j][br] = h[j];
      }
    }
    aPtr += 32;
    bPtr += bStep;
    if (kt + 32 < K) {               // global_prefetch_b8 of next tiles
      __builtin_prefetch(aPtr, 0, 0);
      __builtin_prefetch(bPtr, 0, 0);
    }
    __syncthreads();

    v16h a0 = frag16(&As[wm * 32 + lr][hi * 8]);
    v16h a1 = frag16(&As[wm * 32 + 16 + lr][hi * 8]);
    v16h b0 = frag16(&Bt[wn * 32 + lr][hi * 8]);
    v16h b1 = frag16(&Bt[wn * 32 + 16 + lr][hi * 8]);
    acc00 = __builtin_amdgcn_wmma_f32_16x16x32_f16(false, a0, false, b0, (short)0, acc00, false, false);
    acc01 = __builtin_amdgcn_wmma_f32_16x16x32_f16(false, a0, false, b1, (short)0, acc01, false, false);
    acc10 = __builtin_amdgcn_wmma_f32_16x16x32_f16(false, a1, false, b0, (short)0, acc10, false, false);
    acc11 = __builtin_amdgcn_wmma_f32_16x16x32_f16(false, a1, false, b1, (short)0, acc11, false, false);
    __syncthreads();
  }

  v8f* accs[2][2] = {{&acc00, &acc01}, {&acc10, &acc11}};
#pragma unroll
  for (int i = 0; i < 2; ++i) {
#pragma unroll
    for (int j = 0; j < 2; ++j) {
      v8f a = *accs[i][j];
      int col = bn + wn * 32 + j * 16 + lr;
      float bval = bias ? bias[col] : 0.0f;
#pragma unroll
      for (int r = 0; r < 8; ++r) {
        int row = bm + wm * 32 + i * 16 + r + hi * 8;
        size_t ix = (size_t)row * ldc + col;
        float v = a[r] * alpha + bval;
        if (Res)  v += Res[ix];
        if (relu) v = v > 0.0f ? v : 0.0f;
        C[ix] = v;
      }
    }
  }
}

// -------------------------------------------------------------------------
// Fused flash attention (causal), one block per (64-query tile, b*NHD+h).
// Never materializes scores: S and P tiles live in LDS, online softmax,
// O accumulated in WMMA f32 accumulators with per-row rescale.
// -------------------------------------------------------------------------
__global__ __launch_bounds__(256) void flash_attn(const float* __restrict__ Qg,
                                                  const float* __restrict__ Kg,
                                                  const float* __restrict__ Vg,
                                                  float* __restrict__ Yg) {
  const int bq = blockIdx.x * 64;
  const int z  = blockIdx.y;                 // b*NHD + h
  const int zb = z / NHD, zh = z % NHD;
  const size_t base = (size_t)zb * TP * EXD + (size_t)zh * HDD;
  Qg += base; Kg += base; Vg += base; Yg += base;

  __shared__ alignas(16) _Float16 Qs[64][72];   // [q][d]
  __shared__ alignas(16) _Float16 Kt[64][72];   // [key][d]
  __shared__ alignas(16) _Float16 Vs[64][72];   // [d][key] (transposed)
  __shared__ alignas(16) _Float16 Pf[64][72];   // [q][key]
  __shared__ float Sf[64][68];                  // f32 scores tile
  __shared__ float Mrow[64], Lrow[64], Frow[64];
  __shared__ float pmax[64][4], psum[64][4];

  const int tid = threadIdx.x;
  const int w = tid >> 5, l = tid & 31;
  const int wm = w >> 1, wn = w & 1;           // 4x2 waves, 16x32 of S/O each
  const int lr = l & 15, hi = (l >> 4) & 1;

  {
    int r = tid >> 2, seg = (tid & 3) * 16;
    const float4* p = (const float4*)(Qg + (size_t)(bq + r) * EXD + seg);
    float4 u0 = p[0], u1 = p[1], u2 = p[2], u3 = p[3];
    v8h h0 = {(_Float16)u0.x, (_Float16)u0.y, (_Float16)u0.z, (_Float16)u0.w,
              (_Float16)u1.x, (_Float16)u1.y, (_Float16)u1.z, (_Float16)u1.w};
    v8h h1 = {(_Float16)u2.x, (_Float16)u2.y, (_Float16)u2.z, (_Float16)u2.w,
              (_Float16)u3.x, (_Float16)u3.y, (_Float16)u3.z, (_Float16)u3.w};
    *(v8h*)&Qs[r][seg]     = h0;
    *(v8h*)&Qs[r][seg + 8] = h1;
  }
  if (tid < 64) { Mrow[tid] = -1e30f; Lrow[tid] = 0.0f; }

  v8f o0 = {}, o1 = {};

  for (int kt = 0; kt <= bq; kt += 64) {
    __syncthreads();
    {
      int r = tid >> 2, seg = (tid & 3) * 16;
      const float4* p = (const float4*)(Kg + (size_t)(kt + r) * EXD + seg);
      float4 u0 = p[0], u1 = p[1], u2 = p[2], u3 = p[3];
      v8h h0 = {(_Float16)u0.x, (_Float16)u0.y, (_Float16)u0.z, (_Float16)u0.w,
                (_Float16)u1.x, (_Float16)u1.y, (_Float16)u1.z, (_Float16)u1.w};
      v8h h1 = {(_Float16)u2.x, (_Float16)u2.y, (_Float16)u2.z, (_Float16)u2.w,
                (_Float16)u3.x, (_Float16)u3.y, (_Float16)u3.z, (_Float16)u3.w};
      *(v8h*)&Kt[r][seg]     = h0;
      *(v8h*)&Kt[r][seg + 8] = h1;
    }
    {
      int r = tid >> 3, seg = (tid & 7) * 8;   // key = r, d = seg..seg+7
      const float4* p = (const float4*)(Vg + (size_t)(kt + r) * EXD + seg);
      float4 u0 = p[0], u1 = p[1];
      _Float16 h[8] = {(_Float16)u0.x, (_Float16)u0.y, (_Float16)u0.z, (_Float16)u0.w,
                       (_Float16)u1.x, (_Float16)u1.y, (_Float16)u1.z, (_Float16)u1.w};
#pragma unroll
      for (int j = 0; j < 8; ++j) Vs[seg + j][r] = h[j];
    }
    __syncthreads();

    v8f s0 = {}, s1 = {};
#pragma unroll
    for (int d0 = 0; d0 < 64; d0 += 32) {
      v16h a  = frag16(&Qs[wm * 16 + lr][hi * 8 + d0]);
      v16h b0 = frag16(&Kt[wn * 32 + lr][hi * 8 + d0]);
      v16h b1 = frag16(&Kt[wn * 32 + 16 + lr][hi * 8 + d0]);
      s0 = __builtin_amdgcn_wmma_f32_16x16x32_f16(false, a, false, b0, (short)0, s0, false, false);
      s1 = __builtin_amdgcn_wmma_f32_16x16x32_f16(false, a, false, b1, (short)0, s1, false, false);
    }
#pragma unroll
    for (int r = 0; r < 8; ++r) {
      int row = wm * 16 + r + hi * 8;
      int qg = bq + row;
      int c0 = wn * 32 + lr, c1 = c0 + 16;
      Sf[row][c0] = (kt + c0 <= qg && kt + c0 < TQ) ? s0[r] * 0.125f : -1e30f;
      Sf[row][c1] = (kt + c1 <= qg && kt + c1 < TQ) ? s1[r] * 0.125f : -1e30f;
    }
    __syncthreads();

    int srow = tid >> 2, qu = tid & 3, cb = qu * 16;
    int qg = bq + srow;
    int lim = (qg + 1 < TQ ? qg + 1 : TQ) - kt;        // >= 1 (kt <= bq <= qg)
    float lmax = -1e30f;
#pragma unroll
    for (int j = 0; j < 16; ++j) {
      int c = cb + j;
      if (c < lim) lmax = fmaxf(lmax, Sf[srow][c]);
    }
    pmax[srow][qu] = lmax;
    __syncthreads();
    float mOld = Mrow[srow];
    float newM = fmaxf(fmaxf(fmaxf(pmax[srow][0], pmax[srow][1]),
                             fmaxf(pmax[srow][2], pmax[srow][3])), mOld);
    float fac = __expf(mOld - newM);
    float lsum = 0.0f;
#pragma unroll
    for (int j = 0; j < 16; ++j) {
      int c = cb + j;
      float e = (c < lim) ? __expf(Sf[srow][c] - newM) : 0.0f;
      Pf[srow][c] = (_Float16)e;
      lsum += e;
    }
    psum[srow][qu] = lsum;
    if (qu == 0) Frow[srow] = fac;
    __syncthreads();
    if (qu == 0) {
      Mrow[srow] = newM;
      Lrow[srow] = Lrow[srow] * fac +
                   psum[srow][0] + psum[srow][1] + psum[srow][2] + psum[srow][3];
    }

#pragma unroll
    for (int r = 0; r < 8; ++r) {
      int row = wm * 16 + r + hi * 8;
      float f = Frow[row];
      o0[r] *= f; o1[r] *= f;
    }
#pragma unroll
    for (int k0 = 0; k0 < 64; k0 += 32) {
      v16h a  = frag16(&Pf[wm * 16 + lr][hi * 8 + k0]);
      v16h b0 = frag16(&Vs[wn * 32 + lr][hi * 8 + k0]);
      v16h b1 = frag16(&Vs[wn * 32 + 16 + lr][hi * 8 + k0]);
      o0 = __builtin_amdgcn_wmma_f32_16x16x32_f16(false, a, false, b0, (short)0, o0, false, false);
      o1 = __builtin_amdgcn_wmma_f32_16x16x32_f16(false, a, false, b1, (short)0, o1, false, false);
    }
  }

  __syncthreads();
#pragma unroll
  for (int r = 0; r < 8; ++r) {
    int row = wm * 16 + r + hi * 8;
    float inv = 1.0f / Lrow[row];               // L >= 1 for every row
    int c0 = wn * 32 + lr, c1 = c0 + 16;
    Yg[(size_t)(bq + row) * EXD + c0] = o0[r] * inv;
    Yg[(size_t)(bq + row) * EXD + c1] = o1[r] * inv;
  }
}

// -------------------------------------------------------------------------
// Head, pass 1 (split-K, coalesced): lane = vocab column n, k split over
// NKB blocks. x_flat[b][k] is contiguous (padding lives at k >= 510*768),
// staged to LDS in 256-chunks and broadcast-read. W rows read fully
// coalesced (consecutive lanes -> consecutive addresses).
// -------------------------------------------------------------------------
__global__ __launch_bounds__(256) void head_partial(const float* __restrict__ X,
                                                    const float* __restrict__ W,
                                                    float* __restrict__ P) {
  const int kb = blockIdx.x;           // 0..NKB-1
  const int tid = threadIdx.x;         // vocab column n
  const int k0 = kb * KPB;
  __shared__ float Xs[BQ][256];
  float acc[BQ];
#pragma unroll
  for (int b = 0; b < BQ; ++b) acc[b] = 0.0f;

  for (int t0 = 0; t0 < KPB; t0 += 256) {
    __syncthreads();
#pragma unroll
    for (int b = 0; b < BQ; ++b)
      Xs[b][tid] = X[(size_t)b * TP * EXD + (k0 + t0 + tid)];
    __syncthreads();
    const float* wp = W + (size_t)(k0 + t0) * VOC + tid;
    for (int j = 0; j < 256; ++j) {
      float wv = wp[(size_t)j * VOC];
#pragma unroll
      for (int b = 0; b < BQ; ++b) acc[b] += Xs[b][j] * wv;
    }
  }
  float* Pp = P + (size_t)kb * BQ * VOC + tid;
#pragma unroll
  for (int b = 0; b < BQ; ++b) Pp[(size_t)b * VOC] = acc[b];
}

// Head, pass 2: deterministic fixed-order reduction of NKB partials + bias
__global__ __launch_bounds__(256) void head_reduce(const float* __restrict__ P,
                                                   const float* __restrict__ hb,
                                                   float* __restrict__ out) {
  int b = blockIdx.x, n = threadIdx.x;
  float s = hb[n];
  for (int i = 0; i < NKB; ++i) s += P[((size_t)i * BQ + b) * VOC + n];
  out[b * VOC + n] = s;
}

// -------------------------------------------------------------------------
extern "C" void kernel_launch(void* const* d_in, const int* in_sizes, int n_in,
                              void* d_out, int out_size, void* d_ws, size_t ws_size,
                              hipStream_t stream) {
  (void)in_sizes; (void)n_in; (void)out_size; (void)ws_size;
  const int*   idx    = (const int*)  d_in[0];
  const float* lin_w  = (const float*)d_in[1];
  const float* lin_b  = (const float*)d_in[2];
  const float* ln1_s  = (const float*)d_in[3];
  const float* ln1_b  = (const float*)d_in[4];
  const float* qw     = (const float*)d_in[5];
  const float* qb     = (const float*)d_in[6];
  const float* kw     = (const float*)d_in[7];
  const float* kb     = (const float*)d_in[8];
  const float* vw     = (const float*)d_in[9];
  const float* vb     = (const float*)d_in[10];
  const float* ew     = (const float*)d_in[11];
  const float* eb     = (const float*)d_in[12];
  const float* ln2_s  = (const float*)d_in[13];
  const float* ln2_b  = (const float*)d_in[14];
  const float* m1w    = (const float*)d_in[15];
  const float* m1b    = (const float*)d_in[16];
  const float* m2w    = (const float*)d_in[17];
  const float* m2b    = (const float*)d_in[18];
  const float* lnf_s  = (const float*)d_in[19];
  const float* lnf_b  = (const float*)d_in[20];
  const float* head_w = (const float*)d_in[21];
  const float* head_b = (const float*)d_in[22];
  float* out = (float*)d_out;

  // ---- workspace layout (floats) ----
  float* ws = (float*)d_ws;
  size_t off = 0;
  const size_t ACT = (size_t)ROWS * EXD;
  float* X  = ws + off; off += ACT;                // residual stream [8,512,768]
  float* H  = ws + off; off += ACT;                // LN output (reused)
  float* Q  = ws + off; off += ACT;
  float* Kb = ws + off; off += ACT;
  float* V  = ws + off; off += ACT;
  float* Y  = ws + off; off += ACT;                // attention output
  float* MH = ws + off; off += (size_t)ROWS * HID; // MLP hidden
  float* HP = ws + off; off += (size_t)NKB * BQ * VOC; // head partials

  emb_kernel<<<ROWS, 256, 0, stream>>>(idx, lin_w, lin_b, X);

  dim3 gProj(EXD / 64, ROWS / 128, 1);
  dim3 gM1(HID / 64, ROWS / 128, 1);
  dim3 gFA(TP / 64, BQ * NHD);

  for (int blk = 0; blk < NBLKC; ++blk) {
    const float* Qw = qw + (size_t)blk * EXD * EXD;
    const float* Kw = kw + (size_t)blk * EXD * EXD;
    const float* Vw = vw + (size_t)blk * EXD * EXD;
    const float* Ew = ew + (size_t)blk * EXD * EXD;
    const float* M1 = m1w + (size_t)blk * EXD * HID;
    const float* M2 = m2w + (size_t)blk * HID * EXD;

    // LN1
    ln_kernel<<<ROWS, 256, 0, stream>>>(X, ln1_s + blk * EXD, ln1_b + blk * EXD, H);
    // Q, K, V projections: [4096,768] x [768,768]
    gemm_wmma<false><<<gProj, 256, 0, stream>>>(H, EXD, 0, 0, Qw, EXD, 0, 0,
        qb + blk * EXD, nullptr, 0, 0, Q, EXD, 0, 0, EXD, 1, 1.0f, 0);
    gemm_wmma<false><<<gProj, 256, 0, stream>>>(H, EXD, 0, 0, Kw, EXD, 0, 0,
        kb + blk * EXD, nullptr, 0, 0, Kb, EXD, 0, 0, EXD, 1, 1.0f, 0);
    gemm_wmma<false><<<gProj, 256, 0, stream>>>(H, EXD, 0, 0, Vw, EXD, 0, 0,
        vb + blk * EXD, nullptr, 0, 0, V, EXD, 0, 0, EXD, 1, 1.0f, 0);
    // Fused causal attention: Y = softmax(Q K^T / 8) V  (no scores in HBM)
    flash_attn<<<gFA, 256, 0, stream>>>(Q, Kb, V, Y);
    // Exit projection + residual: X = X + Y @ Ew + eb
    gemm_wmma<false><<<gProj, 256, 0, stream>>>(Y, EXD, 0, 0, Ew, EXD, 0, 0,
        eb + blk * EXD, X, 0, 0, X, EXD, 0, 0, EXD, 1, 1.0f, 0);
    // LN2
    ln_kernel<<<ROWS, 256, 0, stream>>>(X, ln2_s + blk * EXD, ln2_b + blk * EXD, H);
    // MLP: MH = relu(H @ M1 + b1); X = X + MH @ M2 + b2
    gemm_wmma<false><<<gM1, 256, 0, stream>>>(H, EXD, 0, 0, M1, HID, 0, 0,
        m1b + blk * HID, nullptr, 0, 0, MH, HID, 0, 0, EXD, 1, 1.0f, 1);
    gemm_wmma<false><<<gProj, 256, 0, stream>>>(MH, HID, 0, 0, M2, EXD, 0, 0,
        m2b + blk * EXD, X, 0, 0, X, EXD, 0, 0, HID, 1, 1.0f, 0);
  }

  // Final LN + head (split-K coalesced pass + deterministic reduce)
  ln_kernel<<<ROWS, 256, 0, stream>>>(X, lnf_s, lnf_b, H);
  head_partial<<<NKB, 256, 0, stream>>>(H, head_w, HP);
  head_reduce<<<BQ, 256, 0, stream>>>(HP, head_b, out);
}
